// SubjectDicionaryFCNet_7421703488058
// MI455X (gfx1250) — matrix operand order repaired and checked by reading, not compiled, not run on hardware
//
#include <hip/hip_runtime.h>
#include <stdint.h>

typedef __attribute__((ext_vector_type(16))) __bf16 v16bf;
typedef __attribute__((ext_vector_type(8)))  float  v8f;
typedef __attribute__((ext_vector_type(4)))  unsigned u32x4;
typedef __attribute__((ext_vector_type(8)))  int      i32x8;
typedef __attribute__((ext_vector_type(4)))  int      i32x4;

#if defined(__has_builtin)
#if __has_builtin(__builtin_amdgcn_tensor_load_to_lds) && __has_builtin(__builtin_amdgcn_s_wait_tensorcnt)
#define HAS_TDM 1
#endif
#endif

// ---------------- problem constants ----------------
#define BB    4096
#define CCH   22
#define TFULL 1002          // x last-dim (T+1)
#define TT    1001
#define KK    40            // conv output channels
#define KSZ   25
#define PP    100           // pool window
#define WW    977           // conv output width (T-KSZ+1)
#define NWIN  9
#define FEATN 360
#define SSUBJ 9
#define OOUT  4
#define BN_EPS 1e-5f

// ---------------- tiling ----------------
#define WTILE 128           // output positions per workgroup
#define TPS   8             // tiles per sample (ceil(977/128))
#define XROW  160           // LDS row stride == TDM tile_dim0
#define XNEED 152           // WTILE + KSZ - 1 (cols actually read)
#define KRED  550           // C*KSZ flattened reduction
#define KC    18            // ceil(550/32) WMMA K-chunks
#define NTI   3             // N tiles of 16 (40 -> 48)
#define NSLOT 256           // BN-stat contention slots
#define NFRAG (KC*NTI*32*8) // packed B-fragment dwords (13824)

// ---------------- workspace byte offsets ----------------
#define WS_WHI  0
#define WS_WLO  (NFRAG*4)
#define WS_POOL (2*NFRAG*4)
#define WS_STAT (WS_POOL + BB*KK*NWIN*4)
#define WS_COEF (WS_STAT + NSLOT*KK*2*4)

// ---- bf16 hi/lo split helpers (truncation split: f = hi + lo) ----
__device__ __forceinline__ unsigned pack_bf16_hi(float f0, float f1) {
  unsigned u0 = __float_as_uint(f0), u1 = __float_as_uint(f1);
  return (u0 >> 16) | (u1 & 0xFFFF0000u);
}
__device__ __forceinline__ unsigned pack_bf16_lo(float f0, float f1) {
  unsigned u0 = __float_as_uint(f0), u1 = __float_as_uint(f1);
  float h0 = __uint_as_float(u0 & 0xFFFF0000u);
  float h1 = __uint_as_float(u1 & 0xFFFF0000u);
  float r0 = f0 - h0, r1 = f1 - h1;
  return (__float_as_uint(r0) >> 16) | (__float_as_uint(r1) & 0xFFFF0000u);
}

// =============== kernel 0a: zero accumulators ===============
__global__ void zero_kernel(float* p, int n) {
  int i = blockIdx.x * 256 + threadIdx.x;
  if (i < n) p[i] = 0.f;
}

// =============== kernel 0b: pre-pack weight B-fragments ===============
// dword gid = ((chunk*3 + nt)*32 + lane)*8 + i
// B 32x16 bf16 layout: lanes 0-15 -> K rows 0..15 (VGPR i: K=2i,2i+1), lanes 16-31 -> K rows 16..31
__global__ void wfrag_kernel(const float* __restrict__ conv_w,
                             unsigned* __restrict__ wf_hi,
                             unsigned* __restrict__ wf_lo) {
  int gid = blockIdx.x * 256 + threadIdx.x;
  if (gid >= NFRAG) return;
  int i     = gid & 7;
  int lane  = (gid >> 3) & 31;
  int nt    = (gid >> 8) % 3;
  int chunk = gid / (3 * 32 * 8);
  int nl    = lane & 15;
  int hlf   = lane >> 4;
  int k_out = nt * 16 + nl;
  int tg0   = chunk * 32 + hlf * 16 + 2 * i;
  float f0 = 0.f, f1 = 0.f;
  if (k_out < KK) {
    int tg1 = tg0 + 1;
    if (tg0 < KRED) { int c = tg0 / KSZ, t = tg0 - c * KSZ; f0 = conv_w[((size_t)k_out * CCH + c) * KSZ + t]; }
    if (tg1 < KRED) { int c = tg1 / KSZ, t = tg1 - c * KSZ; f1 = conv_w[((size_t)k_out * CCH + c) * KSZ + t]; }
  }
  wf_hi[gid] = pack_bf16_hi(f0, f1);
  wf_lo[gid] = pack_bf16_lo(f0, f1);
}

// ---- B-fragment load + the 3 split-product WMMAs for one N tile ----
__device__ __forceinline__ void do_nt(const unsigned* __restrict__ wf_hi,
                                      const unsigned* __restrict__ wf_lo,
                                      int base, const v16bf& Ahv, const v16bf& Alv, v8f& acc) {
  union { uint4 q[2]; v16bf v; } Bh, Bl;
  const uint4* ph = (const uint4*)(wf_hi + base);
  const uint4* pl = (const uint4*)(wf_lo + base);
  Bh.q[0] = ph[0]; Bh.q[1] = ph[1];
  Bl.q[0] = pl[0]; Bl.q[1] = pl[1];
  acc = __builtin_amdgcn_wmma_f32_16x16x32_bf16(false, Ahv, false, Bh.v, (short)0, acc, false, false);
  acc = __builtin_amdgcn_wmma_f32_16x16x32_bf16(false, Ahv, false, Bl.v, (short)0, acc, false, false);
  acc = __builtin_amdgcn_wmma_f32_16x16x32_bf16(false, Alv, false, Bh.v, (short)0, acc, false, false);
}

// ---- A-fragment build, contiguous k order (k = 16*half + s, as in B) ----
// lanes 0-15: V0-3 K=0..7,  V4-7 K=8..15 ; lanes 16-31: V0-3 K=16..23, V4-7 K=24..31
// MASK=true only needed for the final K-chunk (tg can reach 575 >= KRED)
template <bool MASK>
__device__ __forceinline__ void build_A(const float* __restrict__ ldsx,
                                        const unsigned short* __restrict__ tbl,
                                        int wb, int kb, int tA0, int tA1,
                                        unsigned* ah, unsigned* al) {
#pragma unroll
  for (int i = 0; i < 4; ++i) {
    int tg0 = kb + tA0 + 2 * i;
    int tg1 = tg0 + 1;
    float f0 = ldsx[tbl[tg0] + wb];
    float f1 = ldsx[tbl[tg1] + wb];
    if (MASK) {  // must mask A to exact 0: NaN would propagate through WMMA even vs B=0
      f0 = (tg0 < KRED) ? f0 : 0.f;
      f1 = (tg1 < KRED) ? f1 : 0.f;
    }
    ah[i] = pack_bf16_hi(f0, f1);
    al[i] = pack_bf16_lo(f0, f1);
  }
#pragma unroll
  for (int i = 0; i < 4; ++i) {
    int tg0 = kb + tA1 + 2 * i;
    int tg1 = tg0 + 1;
    float f0 = ldsx[tbl[tg0] + wb];
    float f1 = ldsx[tbl[tg1] + wb];
    if (MASK) {
      f0 = (tg0 < KRED) ? f0 : 0.f;
      f1 = (tg1 < KRED) ? f1 : 0.f;
    }
    ah[4 + i] = pack_bf16_hi(f0, f1);
    al[4 + i] = pack_bf16_lo(f0, f1);
  }
}

// =============== kernel 1: fused implicit-GEMM conv + bias + ELU + BN-stats + avgpool ===============
__global__ __launch_bounds__(256) void eegconv_wmma(
    const float* __restrict__ x,
    const float* __restrict__ conv_b,
    const unsigned* __restrict__ wf_hi,
    const unsigned* __restrict__ wf_lo,
    float* __restrict__ pool,
    float* __restrict__ stat) {
  __shared__ float lds_x[CCH * XROW];          // x tile (f32), row stride 160
  __shared__ float lds_tile[WTILE * 49];       // elu output tile, stride 49 (conflict-free)
  __shared__ float lds_red[2 * 192];
  __shared__ unsigned short off_tbl[KC * 32];  // tg -> (c*XROW + t) LDS offsets

  const int tid  = threadIdx.x;
  const int lane = tid & 31;
  const int wave = tid >> 5;
  const int b    = blockIdx.x / TPS;
  const int tile = blockIdx.x - b * TPS;
  const int w0   = tile * WTILE;

  // -------- stage x[b, :, w0 .. w0+159] into LDS --------
  // TDM path: whole 160-wide tile in-bounds (tiles 0..6). Last tile: manual with zero pad.
  const bool tdm_ok = (w0 + XROW <= TT);
#ifdef HAS_TDM
  if (tdm_ok) {
    if (wave == 0) {
      const unsigned lds_addr = (unsigned)(uintptr_t)(&lds_x[0]);
      const unsigned long long ga = (unsigned long long)(uintptr_t)x +
          ((unsigned long long)b * (CCH * TFULL) + (unsigned long long)w0) * 4ull;
      // D# group0: count=1 | lds_addr | global_addr | type=2
      u32x4 g0 = { 1u, lds_addr, (unsigned)(ga & 0xFFFFFFFFu),
                   (unsigned)(ga >> 32) | 0x80000000u };
      // D# group1: data_size=4B; tensor 160x22 (stride 1002); tile 160x22
      i32x8 g1 = { 0x20000,
                   (int)((XROW & 0xFFFF) << 16),
                   (int)((XROW >> 16) | ((CCH & 0xFFFF) << 16)),
                   (int)((CCH >> 16) | (XROW << 16)),
                   (int)CCH,
                   (int)TFULL,
                   0, 0 };
      i32x4 gz4 = { 0, 0, 0, 0 };
      i32x8 gz8 = { 0, 0, 0, 0, 0, 0, 0, 0 };
      __builtin_amdgcn_tensor_load_to_lds(g0, g1, gz4, gz4, gz8, 0);
    }
  } else
#endif
  {
    for (int idx = tid; idx < CCH * XNEED; idx += 256) {
      const int c = idx / XNEED;
      const int j = idx - c * XNEED;
      const int pos = w0 + j;
      float v = 0.f;
      if (pos < TT) v = x[(size_t)b * (CCH * TFULL) + (size_t)c * TFULL + pos];
      lds_x[c * XROW + j] = v;
    }
  }
  (void)tdm_ok;

  // -------- A-gather offset table (overlaps with staging) --------
  for (int i = tid; i < KC * 32; i += 256) {
    const int c = i / KSZ, t = i - c * KSZ;
    off_tbl[i] = (i < KRED) ? (unsigned short)(c * XROW + t) : (unsigned short)0;
  }
#ifdef HAS_TDM
  __builtin_amdgcn_s_wait_tensorcnt(0);
#endif
  __syncthreads();

  const int m0  = wave * 16;    // wave's M tile within workgroup
  const int nl  = lane & 15;
  const int hlf = lane >> 4;
  const int wb  = m0 + nl;      // A row (output position within tile)
  const int tA0 = hlf ? 16 : 0;
  const int tA1 = hlf ? 24 : 8;

  v8f acc0 = {}; v8f acc1 = {}; v8f acc2 = {};

  // -------- main K loop: chunks 0..16 need no masking (max tg = 543 < 550) --------
  for (int kc = 0; kc < KC - 1; ++kc) {
    union { unsigned u[8]; v16bf v; } Ah, Al;
    build_A<false>(lds_x, off_tbl, wb, kc * 32, tA0, tA1, Ah.u, Al.u);
    __builtin_prefetch(wf_hi + (((kc + 1) * NTI * 32 + lane) << 3), 0, 0);
    __builtin_prefetch(wf_lo + (((kc + 1) * NTI * 32 + lane) << 3), 0, 0);
    do_nt(wf_hi, wf_lo, ((kc * NTI + 0) * 32 + lane) << 3, Ah.v, Al.v, acc0);
    do_nt(wf_hi, wf_lo, ((kc * NTI + 1) * 32 + lane) << 3, Ah.v, Al.v, acc1);
    do_nt(wf_hi, wf_lo, ((kc * NTI + 2) * 32 + lane) << 3, Ah.v, Al.v, acc2);
  }
  // -------- last chunk: mask tg >= 550 to exact zero --------
  {
    const int kc = KC - 1;
    union { unsigned u[8]; v16bf v; } Ah, Al;
    build_A<true>(lds_x, off_tbl, wb, kc * 32, tA0, tA1, Ah.u, Al.u);
    do_nt(wf_hi, wf_lo, ((kc * NTI + 0) * 32 + lane) << 3, Ah.v, Al.v, acc0);
    do_nt(wf_hi, wf_lo, ((kc * NTI + 1) * 32 + lane) << 3, Ah.v, Al.v, acc1);
    do_nt(wf_hi, wf_lo, ((kc * NTI + 2) * 32 + lane) << 3, Ah.v, Al.v, acc2);
  }

  // ---- epilogue: bias + ELU, stage to LDS tile (mask invalid w / k to 0) ----
#define EPI(NT, ACC)                                                              \
  { union { v8f v; float f[8]; } U; U.v = ACC;                                    \
    const int k  = (NT) * 16 + nl;                                                \
    const float bk = conv_b[k < KK ? k : 0];                                      \
    _Pragma("unroll")                                                             \
    for (int r = 0; r < 8; ++r) {                                                 \
      const int m  = hlf * 8 + r;                                                 \
      float v = U.f[r] + bk;                                                      \
      v = v > 0.f ? v : expm1f(v);                                                \
      const int wg = w0 + m0 + m;                                                 \
      const bool ok = (wg < WW) && (k < KK);                                      \
      lds_tile[(m0 + m) * 49 + (NT) * 16 + nl] = ok ? v : 0.f;                    \
    } }
  EPI(0, acc0)
  EPI(1, acc1)
  EPI(2, acc2)
#undef EPI
  __syncthreads();

  // ---- per-channel BN stat partials (sum, sumsq) ----
  if (tid < 192) {
    const int k = tid >> 2, sub = tid & 3;
    float s = 0.f, q = 0.f;
    for (int m = sub; m < WTILE; m += 4) {
      float v = lds_tile[m * 49 + k];
      s += v; q += v * v;
    }
    lds_red[tid] = s;
    lds_red[192 + tid] = q;
  }
  __syncthreads();
  if (tid < KK) {
    float s = lds_red[4*tid] + lds_red[4*tid+1] + lds_red[4*tid+2] + lds_red[4*tid+3];
    float q = lds_red[192+4*tid] + lds_red[192+4*tid+1] + lds_red[192+4*tid+2] + lds_red[192+4*tid+3];
    const int slot = blockIdx.x & (NSLOT - 1);
    atomicAdd(&stat[(slot * KK + tid) * 2 + 0], s);
    atomicAdd(&stat[(slot * KK + tid) * 2 + 1], q);
  }

  // ---- avgpool partial sums (each window receives <= 2 tile contributions) ----
  int win_lo = w0 / PP;
  int win_hi = (w0 + WTILE - 1) / PP;
  if (win_hi > NWIN - 1) win_hi = NWIN - 1;
  const int nwin = win_hi - win_lo + 1;
  if (tid < KK * nwin) {
    const int k   = tid % KK;
    const int win = win_lo + tid / KK;
    int mstart = win * PP - w0;            if (mstart < 0) mstart = 0;
    int mend   = win * PP + PP - 1 - w0;   if (mend > WTILE - 1) mend = WTILE - 1;
    const int mcap = (PP * NWIN - 1) - w0; if (mend > mcap) mend = mcap;
    float s = 0.f;
    for (int m = mstart; m <= mend; ++m) s += lds_tile[m * 49 + k];
    if (mend >= mstart)
      atomicAdd(&pool[((size_t)b * KK + k) * NWIN + win], s);
  }
}

// =============== kernel 2: fold BN into per-channel affine (a_k, b_k) ===============
__global__ void bn_coef_kernel(const float* __restrict__ stat,
                               const float* __restrict__ gamma,
                               const float* __restrict__ beta,
                               float* __restrict__ coef) {
  const int k = threadIdx.x;
  if (k >= KK) return;
  float s = 0.f, q = 0.f;
  for (int i = 0; i < NSLOT; ++i) {
    s += stat[(i * KK + k) * 2 + 0];
    q += stat[(i * KK + k) * 2 + 1];
  }
  const float inv_n = 1.f / ((float)BB * (float)WW);
  const float mean = s * inv_n;
  const float var  = q * inv_n - mean * mean;
  const float a    = gamma[k] * rsqrtf(var + BN_EPS);
  coef[k * 2 + 0] = a;
  coef[k * 2 + 1] = beta[k] - mean * a;
}

// =============== kernel 3: subject-gathered FC (one wave per sample) ===============
__global__ __launch_bounds__(256) void fc_kernel(
    const float* __restrict__ x,
    const float* __restrict__ pool,
    const float* __restrict__ coef,
    const float* __restrict__ fc_w,
    const float* __restrict__ fc_b,
    float* __restrict__ out) {
  const int lane = threadIdx.x & 31;
  const int b    = blockIdx.x * 8 + (threadIdx.x >> 5);
  const float idv = x[(size_t)b * (CCH * TFULL) + (TFULL - 1)];
  int sid = (int)(idv * 1e-6f + 0.5f) - 1;
  sid = sid < 0 ? 0 : (sid > SSUBJ - 1 ? SSUBJ - 1 : sid);
  float a0 = 0.f, a1 = 0.f, a2 = 0.f, a3 = 0.f;
  for (int f = lane; f < FEATN; f += 32) {
    const int k = f / NWIN, n = f - k * NWIN;
    const float feat = coef[2 * k] * pool[((size_t)b * KK + k) * NWIN + n] * (1.f / PP) + coef[2 * k + 1];
    const float* wp = fc_w + ((size_t)sid * FEATN + f) * OOUT;
    a0 += feat * wp[0]; a1 += feat * wp[1]; a2 += feat * wp[2]; a3 += feat * wp[3];
  }
#pragma unroll
  for (int off = 16; off > 0; off >>= 1) {
    a0 += __shfl_down(a0, off, 32);
    a1 += __shfl_down(a1, off, 32);
    a2 += __shfl_down(a2, off, 32);
    a3 += __shfl_down(a3, off, 32);
  }
  if (lane == 0) {
    out[(size_t)b * OOUT + 0] = a0 + fc_b[sid * OOUT + 0];
    out[(size_t)b * OOUT + 1] = a1 + fc_b[sid * OOUT + 1];
    out[(size_t)b * OOUT + 2] = a2 + fc_b[sid * OOUT + 2];
    out[(size_t)b * OOUT + 3] = a3 + fc_b[sid * OOUT + 3];
  }
}

extern "C" void kernel_launch(void* const* d_in, const int* in_sizes, int n_in,
                              void* d_out, int out_size, void* d_ws, size_t ws_size,
                              hipStream_t stream) {
  (void)in_sizes; (void)n_in; (void)out_size; (void)ws_size;
  const float* x      = (const float*)d_in[0];
  const float* conv_w = (const float*)d_in[1];
  const float* conv_b = (const float*)d_in[2];
  const float* bn_g   = (const float*)d_in[3];
  const float* bn_b   = (const float*)d_in[4];
  const float* fc_w   = (const float*)d_in[5];
  const float* fc_b   = (const float*)d_in[6];
  float* out = (float*)d_out;

  char* ws = (char*)d_ws;
  unsigned* wf_hi = (unsigned*)(ws + WS_WHI);
  unsigned* wf_lo = (unsigned*)(ws + WS_WLO);
  float* pool = (float*)(ws + WS_POOL);
  float* stat = (float*)(ws + WS_STAT);
  float* coef = (float*)(ws + WS_COEF);

  // zero pool + stat + coef (contiguous) every launch (graph-replay safe)
  const int nz = BB * KK * NWIN + NSLOT * KK * 2 + KK * 2;
  zero_kernel<<<(nz + 255) / 256, 256, 0, stream>>>(pool, nz);
  wfrag_kernel<<<NFRAG / 256, 256, 0, stream>>>(conv_w, wf_hi, wf_lo);
  eegconv_wmma<<<BB * TPS, 256, 0, stream>>>(x, conv_b, wf_hi, wf_lo, pool, stat);
  bn_coef_kernel<<<1, 64, 0, stream>>>(stat, bn_g, bn_b, coef);
  fc_kernel<<<BB / 8, 256, 0, stream>>>(x, pool, coef, fc_w, fc_b, out);
}
